// DoubleAttention_1511828488686
// MI455X (gfx1250) — compile-verified, matrix operand order checked by reading, not run on hardware
//
#include <hip/hip_runtime.h>
#include <hip/hip_bf16.h>

typedef __bf16 bf16;
typedef __attribute__((ext_vector_type(16))) __bf16 v16bf;
typedef __attribute__((ext_vector_type(8)))  __bf16 v8bf;
typedef __attribute__((ext_vector_type(8)))  float  v8f;

static constexpr int NB = 16;    // batches
static constexpr int C  = 512;   // in/out channels
static constexpr int CM = 128;   // c_m == c_n == 128
static constexpr int NN = 4096;  // h*w

#define WMMA_BF16(a, b, c) \
  __builtin_amdgcn_wmma_f32_16x16x32_bf16(false, (a), false, (b), (short)0, (c), false, false)

// Load one 16x32 bf16 operand fragment from a row-major tile (stride in halves,
// stride must be a multiple of 8 so the b128 loads stay 16B aligned).
// CDNA5 layout: lane&15 = row, (lane>>4)*8 = K base; halves 0..7 = K+0..7,
// halves 8..15 = K+16..23. Works for LDS or global pointers (addrspace inferred).
__device__ __forceinline__ v16bf load_frag(const bf16* p, int stride, int lane) {
  const bf16* r = p + (size_t)(lane & 15) * stride + ((lane >> 4) << 3);
  v8bf lo = *(const v8bf*)(r);
  v8bf hi = *(const v8bf*)(r + 16);
  v16bf f;
#pragma unroll
  for (int i = 0; i < 8; ++i) { f[i] = lo[i]; f[8 + i] = hi[i]; }
  return f;
}

// ---------------------------------------------------------------- K0: weights f32 -> bf16
__global__ __launch_bounds__(256) void k0_convert(const float* __restrict__ wA,
                                                  const float* __restrict__ wB,
                                                  const float* __restrict__ wV,
                                                  const float* __restrict__ wR,
                                                  bf16* __restrict__ Wbf,
                                                  bf16* __restrict__ wRbf) {
  int i = blockIdx.x * 256 + threadIdx.x;
  const int wsz = CM * C;            // 65536 per projection weight
  if (i < 3 * wsz) {
    float v = (i < wsz) ? wA[i] : (i < 2 * wsz ? wB[i - wsz] : wV[i - 2 * wsz]);
    Wbf[i] = (bf16)v;                // stacked row-major [384][512]
  } else {
    int j = i - 3 * wsz;
    if (j < C * CM) wRbf[j] = (bf16)wR[j];
  }
}

__global__ __launch_bounds__(256) void k0_zero(float* __restrict__ gd) {
  int i = blockIdx.x * 256 + threadIdx.x;
  if (i < NB * CM * CM) gd[i] = 0.0f;
}

// ------------------------------------------- K1: fused projections + channel softmax
// grid: (NN/64, NB). Per WG: [384,512] x [512,64] bf16-WMMA GEMM, bias, softmax(B),
// softmax(V), emit A / Bs / Vs as bf16 [b][128][4096].
__global__ __launch_bounds__(256) void k1_proj_softmax(
    const float* __restrict__ x, const bf16* __restrict__ Wbf,
    const float* __restrict__ bA, const float* __restrict__ bB,
    const float* __restrict__ bV,
    bf16* __restrict__ A_ws, bf16* __restrict__ Bs_ws, bf16* __restrict__ Vs_ws) {
  constexpr int XS = 40;   // xT k-stride (halves), multiple of 8
  constexpr int SS = 65;   // score stride (floats), odd -> conflict-free column walks
  __shared__ bf16  xT[64 * XS];      //  5,120 B : x chunk transposed, bf16
  __shared__ float S[384 * SS];      // 99,840 B : GEMM result staging

  const int b    = blockIdx.y;
  const int n0   = blockIdx.x * 64;
  const int tid  = threadIdx.x;
  const int wave = tid >> 5;
  const int lane = tid & 31;
  const float* xb = x + (size_t)b * C * NN;

  v8f acc[3][4] = {};                // wave owns row-tiles 3w..3w+2, 4 col-tiles

  for (int k0 = 0; k0 < C; k0 += 32) {
    // stage x[b][k0..k0+31][n0..n0+63] -> xT[n][k] bf16
    {
      int kk = tid >> 3;             // 0..31
      int nn = (tid & 7) * 8;        // 0,8,..,56
      const float* src = xb + (size_t)(k0 + kk) * NN + n0 + nn;
#pragma unroll
      for (int i = 0; i < 8; ++i) xT[(nn + i) * XS + kk] = (bf16)src[i];
      if (k0 + 32 < C)               // pipeline next chunk into L2 (global_prefetch_b8)
        __builtin_prefetch(src + 32 * NN, 0, 0);
    }
    __syncthreads();
    v16bf afr[3];
#pragma unroll
    for (int j = 0; j < 3; ++j)
      afr[j] = load_frag(Wbf + (size_t)(wave * 48 + j * 16) * C + k0, C, lane);
#pragma unroll
    for (int ct = 0; ct < 4; ++ct) {
      v16bf bfr = load_frag(&xT[ct * 16 * XS], XS, lane);
#pragma unroll
      for (int j = 0; j < 3; ++j) acc[j][ct] = WMMA_BF16(afr[j], bfr, acc[j][ct]);
    }
    __syncthreads();
  }

  // accumulators (+bias) -> LDS score buffer
#pragma unroll
  for (int j = 0; j < 3; ++j)
#pragma unroll
    for (int r = 0; r < 8; ++r) {
      int row = wave * 48 + j * 16 + ((lane >> 4) << 3) + r;  // 0..383
      float bias = (row < 128) ? bA[row] : (row < 256 ? bB[row - 128] : bV[row - 256]);
#pragma unroll
      for (int ct = 0; ct < 4; ++ct)
        S[row * SS + ct * 16 + (lane & 15)] = acc[j][ct][r] + bias;
    }
  __syncthreads();

  // A rows (0..127): straight bf16 writeout, all 256 threads, coalesced columns
  {
    int col = tid & 63;
    bf16* dst = A_ws + (size_t)b * CM * NN + n0 + col;
    for (int r = (tid >> 6); r < CM; r += 4)
      dst[(size_t)r * NN] = (bf16)S[r * SS + col];
  }
  // softmax over the 128-channel axis for B (rows 128..255) and V (rows 256..383)
  if (tid < 128) {
    int col  = tid & 63;
    int base = 128 + (tid >> 6) * 128;
    float m = -INFINITY;
    for (int r = 0; r < CM; ++r) m = fmaxf(m, S[(base + r) * SS + col]);
    float s = 0.0f;
    for (int r = 0; r < CM; ++r) {
      float e = __expf(S[(base + r) * SS + col] - m);
      S[(base + r) * SS + col] = e;
      s += e;
    }
    float inv = 1.0f / s;
    bf16* dst = ((tid >> 6) ? Vs_ws : Bs_ws) + (size_t)b * CM * NN + n0 + col;
    for (int r = 0; r < CM; ++r)
      dst[(size_t)r * NN] = (bf16)(S[(base + r) * SS + col] * inv);
  }
}

// ------------------------------------------------ K2: gd = A · softmax(B)^T  (K=4096)
// grid: (8 K-splits, NB). Per WG: [128,512]x[512,128] partial, atomicAdd into gd f32.
// Staging is a raw bf16 byte copy -> use async global->LDS DMA (ASYNCcnt) with
// double buffering so the copies overlap the WMMA work on the other buffer.
__global__ __launch_bounds__(256) void k2_gd(const bf16* __restrict__ A_ws,
                                             const bf16* __restrict__ Bs_ws,
                                             float* __restrict__ gd) {
  constexpr int KS = 40;             // chunk k-stride (halves); row = 80 B
  __shared__ bf16 Ach[2][128 * KS];  // 2 x 10,240 B
  __shared__ bf16 Bch[2][128 * KS];
  const int b     = blockIdx.y;
  const int nbase = blockIdx.x * (NN / 8);
  const int tid   = threadIdx.x;
  const int wave  = tid >> 5;
  const int lane  = tid & 31;
  const bf16* Ab = A_ws + (size_t)b * CM * NN;
  const bf16* Bb = Bs_ws + (size_t)b * CM * NN;

  // issue async copies of one 128x32-half chunk (A and Bs) into buffer `buf`:
  // 512 16-byte chunks per array, 2 per thread per array.
  auto issue = [&](int kc, int buf) {
#pragma unroll
    for (int h = 0; h < 2; ++h) {
      int cidx = tid + h * 256;          // 0..511
      int row  = cidx >> 2;              // 0..127
      int kk   = (cidx & 3) * 8;         // 0,8,16,24 halves (16B units)
      unsigned lA = (unsigned)(uintptr_t)&Ach[buf][row * KS + kk];
      unsigned lB = (unsigned)(uintptr_t)&Bch[buf][row * KS + kk];
      unsigned long long gA =
          (unsigned long long)(uintptr_t)(Ab + (size_t)row * NN + nbase + kc + kk);
      unsigned long long gB =
          (unsigned long long)(uintptr_t)(Bb + (size_t)row * NN + nbase + kc + kk);
      asm volatile("global_load_async_to_lds_b128 %0, %1, off"
                   :: "v"(lA), "v"(gA) : "memory");
      asm volatile("global_load_async_to_lds_b128 %0, %1, off"
                   :: "v"(lB), "v"(gB) : "memory");
    }
  };

  v8f acc[8] = {};                   // wave owns row-tile `wave`, 8 col-tiles
  issue(0, 0);
  for (int k0 = 0; k0 < NN / 8; k0 += 32) {
    const int cur = (k0 >> 5) & 1;
    asm volatile("s_wait_asynccnt 0x0" ::: "memory");
    __syncthreads();                 // buffer `cur` filled by every wave's DMA
    if (k0 + 32 < NN / 8) issue(k0 + 32, cur ^ 1);  // overlap next chunk's DMA
    v16bf afr = load_frag(&Ach[cur][wave * 16 * KS], KS, lane);
#pragma unroll
    for (int ct = 0; ct < 8; ++ct) {
      v16bf bfr = load_frag(&Bch[cur][ct * 16 * KS], KS, lane);
      acc[ct] = WMMA_BF16(afr, bfr, acc[ct]);
    }
  }
  float* g = gd + (size_t)b * CM * CM;
#pragma unroll
  for (int ct = 0; ct < 8; ++ct)
#pragma unroll
    for (int r = 0; r < 8; ++r) {
      int row = wave * 16 + ((lane >> 4) << 3) + r;
      int col = ct * 16 + (lane & 15);
      atomicAdd(&g[row * CM + col], acc[ct][r]);
    }
}

// ----------------- K3: fused Z = gd · softmax(V), out = wR · Z + bR  (Z stays in LDS)
// grid: (NN/64, NB).
__global__ __launch_bounds__(256) void k3_out(const float* __restrict__ gd,
                                              const bf16* __restrict__ Vs_ws,
                                              const bf16* __restrict__ wRbf,
                                              const float* __restrict__ bR,
                                              float* __restrict__ out) {
  constexpr int GS = 136;            // gd row stride (halves)
  constexpr int TS = 136;            // VsT / ZT row stride (halves)
  __shared__ bf16 gdb[128 * GS];     // 34,816 B
  __shared__ bf16 VsT[64 * TS];      // 17,408 B
  __shared__ bf16 ZT [64 * TS];      // 17,408 B
  const int b    = blockIdx.y;
  const int n0   = blockIdx.x * 64;
  const int tid  = threadIdx.x;
  const int wave = tid >> 5;
  const int lane = tid & 31;

  {  // gd f32 -> bf16 LDS
    const float* g = gd + (size_t)b * CM * CM;
    for (int i = tid; i < CM * CM; i += 256)
      gdb[(i >> 7) * GS + (i & 127)] = (bf16)g[i];
  }
  {  // Vs tile [128 cn][64 n] -> VsT[n][cn]
    const bf16* v = Vs_ws + (size_t)b * CM * NN + n0;
    int row = tid >> 1;
    int nn  = (tid & 1) * 32;
#pragma unroll
    for (int i = 0; i < 32; ++i)
      VsT[(nn + i) * TS + row] = v[(size_t)row * NN + nn + i];
  }
  __syncthreads();

  {  // GEMM1: Z[m,n] = gd · Vs   (wave -> row-tile `wave`, K = 128)
    v8f zacc[4] = {};
    for (int k0 = 0; k0 < CM; k0 += 32) {
      v16bf afr = load_frag(&gdb[wave * 16 * GS + k0], GS, lane);
#pragma unroll
      for (int ct = 0; ct < 4; ++ct) {
        v16bf bfr = load_frag(&VsT[ct * 16 * TS + k0], TS, lane);
        zacc[ct] = WMMA_BF16(afr, bfr, zacc[ct]);
      }
    }
#pragma unroll
    for (int ct = 0; ct < 4; ++ct)
#pragma unroll
      for (int r = 0; r < 8; ++r) {
        int m = wave * 16 + ((lane >> 4) << 3) + r;
        int n = ct * 16 + (lane & 15);
        ZT[n * TS + m] = (bf16)zacc[ct][r];   // store transposed for GEMM2
      }
  }
  __syncthreads();

  // GEMM2: out[c,n] = wR · Z + bR  (wave -> row-tiles 4w..4w+3, K = 128)
  v8f oacc[4][4] = {};
  for (int k0 = 0; k0 < CM; k0 += 32) {
    v16bf bfr[4];
#pragma unroll
    for (int ct = 0; ct < 4; ++ct) bfr[ct] = load_frag(&ZT[ct * 16 * TS + k0], TS, lane);
#pragma unroll
    for (int j = 0; j < 4; ++j) {
      v16bf afr = load_frag(wRbf + (size_t)(wave * 64 + j * 16) * CM + k0, CM, lane);
#pragma unroll
      for (int ct = 0; ct < 4; ++ct) oacc[j][ct] = WMMA_BF16(afr, bfr[ct], oacc[j][ct]);
    }
  }
  float* ob = out + (size_t)b * C * NN;
#pragma unroll
  for (int j = 0; j < 4; ++j)
#pragma unroll
    for (int r = 0; r < 8; ++r) {
      int c = wave * 64 + j * 16 + ((lane >> 4) << 3) + r;
      float bias = bR[c];
#pragma unroll
      for (int ct = 0; ct < 4; ++ct)
        ob[(size_t)c * NN + n0 + ct * 16 + (lane & 15)] = oacc[j][ct][r] + bias;
    }
}

// --------------------------------------------------------------------------- launcher
extern "C" void kernel_launch(void* const* d_in, const int* in_sizes, int n_in,
                              void* d_out, int out_size, void* d_ws, size_t ws_size,
                              hipStream_t stream) {
  const float* x  = (const float*)d_in[0];
  const float* wA = (const float*)d_in[1];
  const float* bA = (const float*)d_in[2];
  const float* wB = (const float*)d_in[3];
  const float* bB = (const float*)d_in[4];
  const float* wV = (const float*)d_in[5];
  const float* bV = (const float*)d_in[6];
  const float* wR = (const float*)d_in[7];
  const float* bR = (const float*)d_in[8];
  float* out = (float*)d_out;

  char* ws = (char*)d_ws;
  bf16* Wbf  = (bf16*)ws;  ws += (size_t)384 * 512 * 2;
  bf16* wRbf = (bf16*)ws;  ws += (size_t)512 * 128 * 2;
  bf16* A_ws = (bf16*)ws;  ws += (size_t)NB * CM * NN * 2;
  bf16* Bs   = (bf16*)ws;  ws += (size_t)NB * CM * NN * 2;
  bf16* Vs   = (bf16*)ws;  ws += (size_t)NB * CM * NN * 2;
  float* gd  = (float*)ws; ws += (size_t)NB * CM * CM * 4;

  k0_convert<<<dim3(1024), dim3(256), 0, stream>>>(wA, wB, wV, wR, Wbf, wRbf);
  k0_zero<<<dim3(1024), dim3(256), 0, stream>>>(gd);
  k1_proj_softmax<<<dim3(NN / 64, NB), dim3(256), 0, stream>>>(x, Wbf, bA, bB, bV,
                                                               A_ws, Bs, Vs);
  k2_gd<<<dim3(8, NB), dim3(256), 0, stream>>>(A_ws, Bs, gd);
  k3_out<<<dim3(NN / 64, NB), dim3(256), 0, stream>>>(gd, Vs, wRbf, bR, out);
}
